// COREAdaptiveSelection_34806414967392
// MI455X (gfx1250) — matrix operand order repaired and checked by reading, not compiled, not run on hardware
//
#include <hip/hip_runtime.h>
#include <hip/hip_bf16.h>

#define NN      8192
#define DD      2048
#define INDIM   3072
#define NCLS    100
#define KSEL    1024
#define CP      112      // padded B rows: 0..99 centroids, 100 = sum(fn), 101..111 zero
#define EPSF    1e-8f
#define NCHUNK  8

typedef __attribute__((ext_vector_type(16))) __bf16 v16bf;
typedef __attribute__((ext_vector_type(8)))  float  v8f;

__device__ __forceinline__ unsigned short f2bf(float f) {
    unsigned int u = __float_as_uint(f);
    unsigned int r = (u + 0x7FFFu + ((u >> 16) & 1u)) >> 16;  // round-to-nearest-even
    return (unsigned short)r;
}

// ---------- k0: zero the accumulation region of the workspace ----------
__global__ void k0_zero(float* __restrict__ z, int n) {
    int i = blockIdx.x * 256 + threadIdx.x;
    if (i < n) z[i] = 0.0f;
}

// ---------- k1: per-row L2 norms + exact float class counts ----------
__global__ void k1_norms(const float* __restrict__ feat, const int* __restrict__ labels,
                         float* __restrict__ norms, float* __restrict__ counts) {
    __shared__ float sred[256];
    int i = blockIdx.x;
    const float* row = feat + (size_t)i * DD;
    float s = 0.0f;
    for (int d = threadIdx.x; d < DD; d += 256) { float v = row[d]; s += v * v; }
    sred[threadIdx.x] = s; __syncthreads();
    for (int off = 128; off > 0; off >>= 1) {
        if (threadIdx.x < off) sred[threadIdx.x] += sred[threadIdx.x + off];
        __syncthreads();
    }
    if (threadIdx.x == 0) {
        norms[i] = sqrtf(sred[0]);
        atomicAdd(&counts[labels[i]], 1.0f);   // integer-valued f32 adds: exact, order-independent
    }
}

// ---------- k2: deterministic partial column-sums of fn and per-class sums ----------
// thread owns column d within its row chunk -> no atomics, bitwise reproducible
__global__ void k2_accum(const float* __restrict__ feat, const int* __restrict__ labels,
                         const float* __restrict__ norms,
                         float* __restrict__ s_part, float* __restrict__ cent_part) {
    int d     = blockIdx.x * 256 + threadIdx.x;   // 0..2047
    int chunk = blockIdx.y;                        // 0..7
    int r0    = chunk * (NN / NCHUNK);
    float* cp = cent_part + (size_t)chunk * NCLS * DD + d;
    float ssum = 0.0f;
    for (int r = r0; r < r0 + NN / NCHUNK; ++r) {
        float invn = 1.0f / fmaxf(norms[r], EPSF);
        float v = feat[(size_t)r * DD + d] * invn;
        ssum += v;
        cp[(size_t)labels[r] * DD] += v;           // column owned by this thread
    }
    s_part[(size_t)chunk * DD + d] = ssum;
}

// ---------- k2b: fold the NCHUNK partials ----------
__global__ void k2b_reduce(const float* __restrict__ s_part, const float* __restrict__ cent_part,
                           float* __restrict__ s_vec, float* __restrict__ cent_sum) {
    int idx = blockIdx.x * 256 + threadIdx.x;
    const int total = DD + NCLS * DD;
    if (idx >= total) return;
    float acc = 0.0f;
    if (idx < DD) {
        for (int c = 0; c < NCHUNK; ++c) acc += s_part[c * DD + idx];
        s_vec[idx] = acc;
    } else {
        int j = idx - DD;
        for (int c = 0; c < NCHUNK; ++c) acc += cent_part[(size_t)c * NCLS * DD + j];
        cent_sum[j] = acc;
    }
}

// ---------- k3: build bf16 B matrix [CP x DD]: unit centroids, sum(fn), zero pad ----------
__global__ void k3_buildB(const float* __restrict__ cent_sum, const float* __restrict__ counts,
                          const float* __restrict__ s_vec, unsigned short* __restrict__ bmat) {
    __shared__ float sred[256];
    int n = blockIdx.x;                 // 0..111
    unsigned short* out = bmat + (size_t)n * DD;
    if (n > NCLS) { for (int d = threadIdx.x; d < DD; d += 256) out[d] = 0; return; }
    if (n == NCLS) { for (int d = threadIdx.x; d < DD; d += 256) out[d] = f2bf(s_vec[d]); return; }
    float cnt = fmaxf(counts[n], 1.0f);
    const float* cs = cent_sum + (size_t)n * DD;
    float s = 0.0f;
    for (int d = threadIdx.x; d < DD; d += 256) { float v = cs[d] / cnt; s += v * v; }
    sred[threadIdx.x] = s; __syncthreads();
    for (int off = 128; off > 0; off >>= 1) {
        if (threadIdx.x < off) sred[threadIdx.x] += sred[threadIdx.x + off];
        __syncthreads();
    }
    float scale = 1.0f / (fmaxf(sqrtf(sred[0]), EPSF) * cnt);
    for (int d = threadIdx.x; d < DD; d += 256) out[d] = f2bf(cs[d] * scale);
}

// ---------- k4: WMMA GEMM  cos[8192 x 112] = fn @ B^T  (bf16 -> f32) ----------
// 8 waves/block, each wave owns one 16-row M tile and all 7 N tiles (A read once).
__global__ void __launch_bounds__(256)
k4_gemm(const float* __restrict__ feat, const float* __restrict__ norms,
        const unsigned short* __restrict__ bmat, float* __restrict__ cosf) {
    const int lane = threadIdx.x & 31;
    const int wave = threadIdx.x >> 5;
    const int m = lane & 15;            // row-in-tile (A) / col-in-tile (B, C)
    const int h = lane >> 4;            // lane half selects K sub-range
    const int row_tile = (blockIdx.x * 8 + wave) * 16;
    const int row = row_tile + m;
    const float invn = 1.0f / fmaxf(norms[row], EPSF);
    const float* arow = feat + (size_t)row * DD;

    v8f acc[7];
#pragma unroll
    for (int t = 0; t < 7; ++t) acc[t] = (v8f){0.f,0.f,0.f,0.f,0.f,0.f,0.f,0.f};

    union { v16bf v; unsigned short u[16]; } A;
    union { v16bf v; uint4 q[2];          } B;

    for (int k0 = 0; k0 < DD; k0 += 32) {
        // A layout (bf16 16x32): lane half h holds K = k0+8h+{0..7} and k0+16+8h+{0..7}
        const float* a0 = arow + k0 + h * 8;
        __builtin_prefetch(arow + k0 + 32, 0, 0);   // global_prefetch for next K slab
        float4 f0 = *(const float4*)(a0);
        float4 f1 = *(const float4*)(a0 + 4);
        float4 f2 = *(const float4*)(a0 + 16);
        float4 f3 = *(const float4*)(a0 + 20);
        A.u[0]=f2bf(f0.x*invn);  A.u[1]=f2bf(f0.y*invn);  A.u[2]=f2bf(f0.z*invn);  A.u[3]=f2bf(f0.w*invn);
        A.u[4]=f2bf(f1.x*invn);  A.u[5]=f2bf(f1.y*invn);  A.u[6]=f2bf(f1.z*invn);  A.u[7]=f2bf(f1.w*invn);
        A.u[8]=f2bf(f2.x*invn);  A.u[9]=f2bf(f2.y*invn);  A.u[10]=f2bf(f2.z*invn); A.u[11]=f2bf(f2.w*invn);
        A.u[12]=f2bf(f3.x*invn); A.u[13]=f2bf(f3.y*invn); A.u[14]=f2bf(f3.z*invn); A.u[15]=f2bf(f3.w*invn);
#pragma unroll
        for (int t = 0; t < 7; ++t) {
            // B layout (bf16 32x16): lane = N col, half h holds K = k0+8h+{0..7} and k0+16+8h+{0..7}
            const unsigned short* bp = bmat + (size_t)(t * 16 + m) * DD + k0 + h * 8;
            B.q[0] = *(const uint4*)(bp);
            B.q[1] = *(const uint4*)(bp + 16);
            acc[t] = __builtin_amdgcn_wmma_f32_16x16x32_bf16(
                false, A.v, false, B.v, (short)0, acc[t], false, false);
        }
    }
    // C layout: VGPR r -> M = 8h + r, N = lane&15
#pragma unroll
    for (int t = 0; t < 7; ++t)
#pragma unroll
        for (int r = 0; r < 8; ++r)
            cosf[(size_t)(row_tile + h * 8 + r) * CP + t * 16 + m] = acc[t][r];
}

// ---------- k5: quality = 0.7*cos_cent + 0.3*(1 - (rowsum - selfdot)/(N-1)) ----------
__global__ void k5_quality(const float* __restrict__ cosf, const int* __restrict__ labels,
                           const float* __restrict__ norms, float* __restrict__ quality) {
    int i = blockIdx.x * 256 + threadIdx.x;
    if (i >= NN) return;
    float cosc   = cosf[(size_t)i * CP + labels[i]];
    float rowsum = cosf[(size_t)i * CP + NCLS];          // fn_i . sum_j fn_j
    float r = norms[i] / fmaxf(norms[i], EPSF);
    float selfdot = r * r;                               // S[i,i]
    float diversity = 1.0f - (rowsum - selfdot) / (float)(NN - 1);
    quality[i] = 0.7f * cosc + 0.3f * diversity;
}

// ---------- k6: exact top-k by rank counting (jax tie rule: lower index first) ----------
__global__ void k6_rank(const float* __restrict__ q, int* __restrict__ sel) {
    __shared__ int sred[256];
    int i = blockIdx.x;
    float qi = q[i];
    int cnt = 0;
    for (int j = threadIdx.x; j < NN; j += 256) {
        float qj = q[j];
        cnt += (qj > qi) || (qj == qi && j < i);
    }
    sred[threadIdx.x] = cnt; __syncthreads();
    for (int off = 128; off > 0; off >>= 1) {
        if (threadIdx.x < off) sred[threadIdx.x] += sred[threadIdx.x + off];
        __syncthreads();
    }
    if (threadIdx.x == 0 && sred[0] < KSEL) sel[sred[0]] = i;
}

// ---------- k7: gather selected inputs + labels(as float) into d_out ----------
__global__ void k7_gather(const float* __restrict__ inputs, const int* __restrict__ labels,
                          const int* __restrict__ sel, float* __restrict__ out) {
    int b = blockIdx.x;
    int src = sel[b];
    const float* srow = inputs + (size_t)src * INDIM;
    float* drow = out + (size_t)b * INDIM;
    for (int d = threadIdx.x; d < INDIM; d += 256) drow[d] = srow[d];
    if (threadIdx.x == 0) out[(size_t)KSEL * INDIM + b] = (float)labels[src];
}

extern "C" void kernel_launch(void* const* d_in, const int* in_sizes, int n_in,
                              void* d_out, int out_size, void* d_ws, size_t ws_size,
                              hipStream_t stream) {
    const float* inputs = (const float*)d_in[0];
    const int*   labels = (const int*)d_in[1];
    const float* feat   = (const float*)d_in[2];
    float* out = (float*)d_out;

    // workspace layout (all 16B aligned by construction)
    float* w         = (float*)d_ws;
    float* norms     = w;                                        // NN
    float* counts    = norms + NN;                               // NCLS   (zeroed)
    float* s_vec     = counts + NCLS;                            // DD     (zeroed)
    float* cent_sum  = s_vec + DD;                               // NCLS*DD(zeroed)
    float* s_part    = cent_sum + (size_t)NCLS * DD;             // NCHUNK*DD (zeroed)
    float* cent_part = s_part + (size_t)NCHUNK * DD;             // NCHUNK*NCLS*DD (zeroed)
    unsigned short* bmat = (unsigned short*)(cent_part + (size_t)NCHUNK * NCLS * DD); // CP*DD
    float* cosf      = (float*)(bmat + (size_t)CP * DD);         // NN*CP
    float* quality   = cosf + (size_t)NN * CP;                   // NN
    int*   sel       = (int*)(quality + NN);                     // KSEL

    const int zero_n = NCLS + DD + NCLS * DD + NCHUNK * DD + NCHUNK * NCLS * DD;

    k0_zero   <<<(zero_n + 255) / 256, 256, 0, stream>>>(counts, zero_n);
    k1_norms  <<<NN, 256, 0, stream>>>(feat, labels, norms, counts);
    k2_accum  <<<dim3(DD / 256, NCHUNK), 256, 0, stream>>>(feat, labels, norms, s_part, cent_part);
    k2b_reduce<<<((DD + NCLS * DD) + 255) / 256, 256, 0, stream>>>(s_part, cent_part, s_vec, cent_sum);
    k3_buildB <<<CP, 256, 0, stream>>>(cent_sum, counts, s_vec, bmat);
    k4_gemm   <<<NN / 128, 256, 0, stream>>>(feat, norms, bmat, cosf);
    k5_quality<<<(NN + 255) / 256, 256, 0, stream>>>(cosf, labels, norms, quality);
    k6_rank   <<<NN, 256, 0, stream>>>(quality, sel);
    k7_gather <<<KSEL, 256, 0, stream>>>(inputs, labels, sel, out);
}
